// ContextualBilateralLoss_30305289240834
// MI455X (gfx1250) — matrix operand order repaired and checked by reading, not compiled, not run on hardware
//
#include <hip/hip_runtime.h>

// ---------------------------------------------------------------------------
// Types for CDNA5 WMMA (wave32)
// ---------------------------------------------------------------------------
typedef __bf16 v16bf __attribute__((ext_vector_type(16)));
typedef float  v8f   __attribute__((ext_vector_type(8)));

__device__ __forceinline__ unsigned short f2bf(float f) {
    unsigned int u = __float_as_uint(f);
    unsigned int r = (u + 0x7FFFu + ((u >> 16) & 1u)) >> 16;   // round-to-nearest-even
    return (unsigned short)r;
}
__device__ __forceinline__ float bf2f(unsigned short h) {
    return __uint_as_float(((unsigned int)h) << 16);
}

// Swizzled LDS index helpers: element (row, k) goes to the (lane, slot) that the
// 16-bit WMMA operand layout expects, so a lane's fragment is 16 contiguous bf16
// (32B -> ds_load_b128 x2).
// A-matrix 16x32 (ISA 7.12.2): lane = m + 16*((k>>3)&1), slot = 8*(k>>4) + (k&7)
__device__ __forceinline__ int swzA(int ml, int kl) {
    int sub  = ml >> 4;
    int lane = (ml & 15) + (((kl >> 3) & 1) << 4);
    int slot = ((kl >> 4) << 3) | (kl & 7);
    return (((sub << 5) + lane) << 4) + slot;
}
// B-matrix 32x16: lane = n + 16*((k>>3)&1), slot = 8*(k>>4) + (k&7)
__device__ __forceinline__ int swzB(int nl, int kl) {
    int sub  = nl >> 4;
    int lane = (nl & 15) + (((kl >> 3) & 1) << 4);
    int slot = ((kl >> 4) << 3) | (kl & 7);
    return (((sub << 5) + lane) << 4) + slot;
}

// ---------------------------------------------------------------------------
// WMMA GEMM: C[M][N] = A[M][Kpad] (bf16 row-major, zero-padded in K) * B (bf16)
//   CONV = 1 : B = implicit im2col of activations act[Cin][convH][convW]
//   CONV = 0 : B = plain row-major K x N
// Block: 256 threads = 8 waves; block tile 64(M) x 128(N); K step 32.
// Wave w: msub = w&3; n-quad = (w>>2)*4 -> 4 WMMAs per k-step sharing one A frag.
// Pipeline per k-step: fragment ds_loads -> next-tile global loads -> WMMAs
// (wait DScnt only) -> next-tile ds_stores (wait LOADcnt) -> barrier.
// Requirements (caller): M % 64 == 0, N % 128 == 0, Kpad % 32 == 0.
// ---------------------------------------------------------------------------
template <int CONV>
__global__ __launch_bounds__(256) void wmma_gemm_kernel(
    const unsigned short* __restrict__ A,
    const unsigned short* __restrict__ Bsrc,
    const float* __restrict__ bias,
    float* __restrict__ outF,
    unsigned short* __restrict__ outB,
    int M, int N, int K, int Kpad,
    int convH, int convW, int relu)
{
    __shared__ __align__(32) unsigned short As[2][2048];   // 4 msub x 32 lanes x 16
    __shared__ __align__(32) unsigned short Bs[2][4096];   // 8 nsub x 32 lanes x 16

    const int t    = threadIdx.x;
    const int lane = t & 31;
    const int wv   = t >> 5;
    const int msub = wv & 3;
    const int nq   = (wv >> 2) << 2;      // 0 or 4: first of this wave's 4 n-subtiles

    const int n0 = blockIdx.x << 7;
    const int m0 = blockIdx.y << 6;

    v8f acc0 = {}, acc1 = {}, acc2 = {}, acc3 = {};

    // ---- per-thread B-staging geometry, constant across the K loop ----
    // B tile is 32(K) x 128(N): e = (r<<8)+t, kl = e>>7, nl = e&127.
    int gnArr[16];
    int flArr[16];
    const int HW = convH * convW;
#pragma unroll
    for (int r = 0; r < 16; ++r) {
        int e  = (r << 8) + t;
        int nl = e & 127;
        int gn = n0 + nl;
        gnArr[r] = gn;
        if (CONV) {
            int oh = gn / convW;
            int ow = gn - oh * convW;
            flArr[r] = (oh > 0 ? 1 : 0) | (oh < convH - 1 ? 2 : 0) |
                       (ow > 0 ? 4 : 0) | (ow < convW - 1 ? 8 : 0);
        } else {
            flArr[r] = 0;
        }
    }

    // ---- staging phases (branchless; load phase first, store phase later) ----
    auto loadA = [&](int k0, unsigned int (&aval)[4]) {
#pragma unroll
        for (int r = 0; r < 4; ++r) {
            int p  = (r << 8) + t;          // pair index 0..1023
            int ml = p >> 4;
            int kp = (p & 15) << 1;
            aval[r] = *reinterpret_cast<const unsigned int*>(
                &A[(size_t)(m0 + ml) * Kpad + (k0 + kp)]);
        }
    };
    auto loadB = [&](int k0, unsigned short (&bval)[16]) {
#pragma unroll
        for (int r = 0; r < 16; ++r) {
            int e  = (r << 8) + t;
            int kl = e >> 7;
            int gk = k0 + kl;
            if (CONV) {
                int fl  = flArr[r];
                int cin = gk / 9;
                int tap = gk - cin * 9;
                int dr  = tap / 3;                 // 0..2 (offset -1)
                int dc  = tap - dr * 3;            // 0..2 (offset -1)
                bool okh = (dr == 1) || (dr == 0 ? (fl & 1) != 0 : (fl & 2) != 0);
                bool okw = (dc == 1) || (dc == 0 ? (fl & 4) != 0 : (fl & 8) != 0);
                bool ok  = (gk < K) && okh && okw;
                long idx = (long)cin * HW + gnArr[r] + (dr - 1) * convW + (dc - 1);
                idx = ok ? idx : 0;
                unsigned short v = Bsrc[idx];
                bval[r] = ok ? v : (unsigned short)0;
            } else {
                bool ok = (gk < K);
                size_t idx = ok ? ((size_t)gk * N + gnArr[r]) : 0;
                unsigned short v = Bsrc[idx];
                bval[r] = ok ? v : (unsigned short)0;
            }
        }
    };
    auto storeAB = [&](int buf, const unsigned int (&aval)[4],
                       const unsigned short (&bval)[16]) {
#pragma unroll
        for (int r = 0; r < 4; ++r) {
            int p  = (r << 8) + t;
            int ml = p >> 4;
            int kp = (p & 15) << 1;
            *reinterpret_cast<unsigned int*>(&As[buf][swzA(ml, kp)]) = aval[r];
        }
#pragma unroll
        for (int r = 0; r < 16; ++r) {
            int e  = (r << 8) + t;
            int kl = e >> 7;
            int nl = e & 127;
            Bs[buf][swzB(nl, kl)] = bval[r];
        }
    };

    const int nk = Kpad >> 5;
    {
        unsigned int aval[4]; unsigned short bval[16];
        loadA(0, aval);
        loadB(0, bval);
        storeAB(0, aval, bval);
    }
    __syncthreads();

    for (int ks = 0; ks < nk; ++ks) {
        const int cur = ks & 1;

        // 1) fragment loads from current buffer (distinct registers)
        const v16bf a  = *reinterpret_cast<const v16bf*>(
                             &As[cur][((msub << 5) + lane) << 4]);
        const v16bf b0 = *reinterpret_cast<const v16bf*>(
                             &Bs[cur][(((nq + 0) << 5) + lane) << 4]);
        const v16bf b1 = *reinterpret_cast<const v16bf*>(
                             &Bs[cur][(((nq + 1) << 5) + lane) << 4]);
        const v16bf b2 = *reinterpret_cast<const v16bf*>(
                             &Bs[cur][(((nq + 2) << 5) + lane) << 4]);
        const v16bf b3 = *reinterpret_cast<const v16bf*>(
                             &Bs[cur][(((nq + 3) << 5) + lane) << 4]);

        // 2) next-tile global loads (overlap HBM latency with WMMA below)
        unsigned int aval[4]; unsigned short bval[16];
        const bool have = (ks + 1 < nk);
        if (have) {
            loadA((ks + 1) << 5, aval);
            loadB((ks + 1) << 5, bval);
        }

        // 3) matrix math (waits only on the fragment ds_loads)
        acc0 = __builtin_amdgcn_wmma_f32_16x16x32_bf16(
                   false, a, false, b0, (short)0, acc0, false, false);
        acc1 = __builtin_amdgcn_wmma_f32_16x16x32_bf16(
                   false, a, false, b1, (short)0, acc1, false, false);
        acc2 = __builtin_amdgcn_wmma_f32_16x16x32_bf16(
                   false, a, false, b2, (short)0, acc2, false, false);
        acc3 = __builtin_amdgcn_wmma_f32_16x16x32_bf16(
                   false, a, false, b3, (short)0, acc3, false, false);

        // 4) next-tile LDS stores (wait LOADcnt; after WMMAs so DScnt stays clean)
        if (have) storeAB(cur ^ 1, aval, bval);

        __syncthreads();
    }

    // ---- epilogue (no M/N guards: exact tiling) ----
    // C layout (ISA 7.12.2): m = 8*(lane>=16) + vgpr, n = lane&15
    const int mbase = m0 + (msub << 4) + ((lane >> 4) << 3);
    const int ncb   = n0 + (nq << 4) + (lane & 15);
#pragma unroll
    for (int v = 0; v < 8; ++v) {
        int m = mbase + v;
        float bv = bias ? bias[m] : 0.0f;
        size_t rowo = (size_t)m * N;
        float x0 = acc0[v] + bv, x1 = acc1[v] + bv, x2 = acc2[v] + bv, x3 = acc3[v] + bv;
        if (relu) {
            x0 = fmaxf(x0, 0.0f); x1 = fmaxf(x1, 0.0f);
            x2 = fmaxf(x2, 0.0f); x3 = fmaxf(x3, 0.0f);
        }
        size_t i0 = rowo + (size_t)ncb;
        if (outF) {
            outF[i0] = x0; outF[i0 + 16] = x1; outF[i0 + 32] = x2; outF[i0 + 48] = x3;
        }
        if (outB) {
            outB[i0] = f2bf(x0); outB[i0 + 16] = f2bf(x1);
            outB[i0 + 32] = f2bf(x2); outB[i0 + 48] = f2bf(x3);
        }
    }
}

// ---------------------------------------------------------------------------
// Small support kernels
// ---------------------------------------------------------------------------
__global__ void convert_weights_kernel(const float* __restrict__ src,
                                       unsigned short* __restrict__ dst,
                                       int M, int K, int Kpad) {
    int i = blockIdx.x * 256 + threadIdx.x;
    if (i >= M * Kpad) return;
    int m = i / Kpad, k = i - m * Kpad;
    dst[i] = (k < K) ? f2bf(src[(size_t)m * K + k]) : (unsigned short)0;
}

__global__ void preprocess_kernel(const float* __restrict__ img,
                                  unsigned short* __restrict__ out, int n) {
    int i = blockIdx.x * 256 + threadIdx.x;
    if (i >= n) return;
    int c = i >> 16;   // 256*256 = 65536 per channel
    float mean = (c == 0) ? 0.485f : ((c == 1) ? 0.456f : 0.406f);
    float inv  = (c == 0) ? (1.0f / 0.229f) : ((c == 1) ? (1.0f / 0.224f) : (1.0f / 0.225f));
    out[i] = f2bf((img[i] - mean) * inv);
}

__global__ void maxpool_kernel(const unsigned short* __restrict__ in,
                               unsigned short* __restrict__ out,
                               int C, int OH, int OW) {
    int i = blockIdx.x * 256 + threadIdx.x;
    int tot = C * OH * OW;
    if (i >= tot) return;
    int ow = i % OW;
    int tmp = i / OW;
    int oh = tmp % OH;
    int c  = tmp / OH;
    int IW = OW * 2, IH = OH * 2;
    const unsigned short* p = in + ((size_t)c * IH + oh * 2) * IW + ow * 2;
    float a = bf2f(p[0]), b = bf2f(p[1]), d = bf2f(p[IW]), e = bf2f(p[IW + 1]);
    out[i] = f2bf(fmaxf(fmaxf(a, b), fmaxf(d, e)));
}

__global__ void channel_mean_kernel(const float* __restrict__ f,
                                    float* __restrict__ mu) {
    int c = blockIdx.x, t = threadIdx.x;
    float s = 0.0f;
    for (int i = t; i < 4096; i += 256) s += f[(size_t)c * 4096 + i];
    __shared__ float red[256];
    red[t] = s; __syncthreads();
    for (int k = 128; k > 0; k >>= 1) {
        if (t < k) red[t] += red[t + k];
        __syncthreads();
    }
    if (t == 0) mu[c] = red[0] * (1.0f / 4096.0f);
}

// layoutA=1: out[i*256 + c] (HW x C, GEMM A) ; layoutA=0: out[c*4096 + i] (C x HW, GEMM B)
__global__ void normalize_kernel(const float* __restrict__ f,
                                 const float* __restrict__ mu,
                                 unsigned short* __restrict__ out, int layoutA) {
    int i = blockIdx.x, c = threadIdx.x;     // 256 channels
    float v = f[(size_t)c * 4096 + i] - mu[c];
    __shared__ float red[256];
    red[c] = v * v; __syncthreads();
    for (int k = 128; k > 0; k >>= 1) {
        if (c < k) red[c] += red[c + k];
        __syncthreads();
    }
    float inv = 1.0f / fmaxf(sqrtf(red[0]), 1e-12f);
    unsigned short b = f2bf(v * inv);
    if (layoutA) out[(size_t)i * 256 + c] = b;
    else         out[(size_t)c * 4096 + i] = b;
}

// Per-row stats over the 4096x4096 cos matrix + analytic spatial distances.
// stats layout: [0:4096)=min_feat, [4096)=min_sp, [8192)=wsum_feat, [12288)=wsum_sp
__global__ void cx_rowstats_kernel(const float* __restrict__ cosm,
                                   float* __restrict__ stats) {
    const int i = blockIdx.x, t = threadIdx.x;
    const float inv65 = 1.0f / 65.0f;
    const float gri = (float)(i >> 6) * inv65, gci = (float)(i & 63) * inv65;
    const float si  = gri * gri + gci * gci;
    const float* row = cosm + (size_t)i * 4096;

    float mnf = 3.4e38f, mns = 3.4e38f;
    for (int j = t; j < 4096; j += 256) {
        float df = 1.0f - row[j];
        float grj = (float)(j >> 6) * inv65, gcj = (float)(j & 63) * inv65;
        float ds = si + grj * grj + gcj * gcj - 2.0f * (gri * grj + gci * gcj);
        ds = fmaxf(ds, 0.0f);
        mnf = fminf(mnf, df);
        mns = fminf(mns, ds);
    }
    __shared__ float r0[256], r1[256];
    r0[t] = mnf; r1[t] = mns; __syncthreads();
    for (int k = 128; k > 0; k >>= 1) {
        if (t < k) { r0[t] = fminf(r0[t], r0[t + k]); r1[t] = fminf(r1[t], r1[t + k]); }
        __syncthreads();
    }
    const float minf = r0[0], mins = r1[0];
    const float invf = 1.0f / (minf + 1e-5f);
    const float invs = 1.0f / (mins + 1e-5f);
    __syncthreads();

    float sf = 0.0f, ss = 0.0f;
    for (int j = t; j < 4096; j += 256) {
        float df = 1.0f - row[j];
        float grj = (float)(j >> 6) * inv65, gcj = (float)(j & 63) * inv65;
        float ds = si + grj * grj + gcj * gcj - 2.0f * (gri * grj + gci * gcj);
        ds = fmaxf(ds, 0.0f);
        sf += __expf((1.0f - df * invf) * 2.0f);   // band_width = 0.5 -> x2
        ss += __expf((1.0f - ds * invs) * 2.0f);
    }
    r0[t] = sf; r1[t] = ss; __syncthreads();
    for (int k = 128; k > 0; k >>= 1) {
        if (t < k) { r0[t] += r0[t + k]; r1[t] += r1[t + k]; }
        __syncthreads();
    }
    if (t == 0) {
        stats[i]         = minf;
        stats[4096 + i]  = mins;
        stats[8192 + i]  = r0[0];
        stats[12288 + i] = r1[0];
    }
}

__global__ void cx_rowmax_kernel(const float* __restrict__ cosm,
                                 const float* __restrict__ stats,
                                 float* __restrict__ kmax) {
    const int i = blockIdx.x, t = threadIdx.x;
    const float inv65 = 1.0f / 65.0f;
    const float gri = (float)(i >> 6) * inv65, gci = (float)(i & 63) * inv65;
    const float si  = gri * gri + gci * gci;
    const float* row = cosm + (size_t)i * 4096;

    const float invf  = 1.0f / (stats[i] + 1e-5f);
    const float invs  = 1.0f / (stats[4096 + i] + 1e-5f);
    const float iwf   = 0.5f / stats[8192 + i];    // (1 - weight_sp) = 0.5
    const float iws   = 0.5f / stats[12288 + i];   // weight_sp = 0.5

    float mx = -3.4e38f;
    for (int j = t; j < 4096; j += 256) {
        float df = 1.0f - row[j];
        float grj = (float)(j >> 6) * inv65, gcj = (float)(j & 63) * inv65;
        float ds = si + grj * grj + gcj * gcj - 2.0f * (gri * grj + gci * gcj);
        ds = fmaxf(ds, 0.0f);
        float cxf = __expf((1.0f - df * invf) * 2.0f) * iwf;
        float cxs = __expf((1.0f - ds * invs) * 2.0f) * iws;
        mx = fmaxf(mx, cxf + cxs);
    }
    __shared__ float red[256];
    red[t] = mx; __syncthreads();
    for (int k = 128; k > 0; k >>= 1) {
        if (t < k) red[t] = fmaxf(red[t], red[t + k]);
        __syncthreads();
    }
    if (t == 0) kmax[i] = red[0];
}

__global__ void cx_final_kernel(const float* __restrict__ kmax,
                                float* __restrict__ out) {
    int t = threadIdx.x;
    float s = 0.0f;
    for (int i = t; i < 4096; i += 256) s += kmax[i];
    __shared__ float red[256];
    red[t] = s; __syncthreads();
    for (int k = 128; k > 0; k >>= 1) {
        if (t < k) red[t] += red[t + k];
        __syncthreads();
    }
    if (t == 0) {
        float cx = red[0] * (1.0f / 4096.0f);
        out[0] = -logf(cx + 1e-5f);
    }
}

// ---------------------------------------------------------------------------
// Host-side orchestration
// ---------------------------------------------------------------------------
static const int CIN [8] = {  3, 64,  64, 128, 128, 256, 256, 256};
static const int COUT[8] = { 64, 64, 128, 128, 256, 256, 256, 256};
static const int SIDE[8] = {256,256, 128, 128,  64,  64,  64,  64};

extern "C" void kernel_launch(void* const* d_in, const int* in_sizes, int n_in,
                              void* d_out, int out_size, void* d_ws, size_t ws_size,
                              hipStream_t stream) {
    (void)in_sizes; (void)n_in; (void)out_size; (void)ws_size;

    // ---- carve workspace ----
    char* W = (char*)d_ws;
    size_t off = 0;
    auto carve = [&](size_t bytes) -> void* {
        void* p = W + off;
        off += (bytes + 255) & ~(size_t)255;
        return p;
    };
    int Kreal[8], Kpad[8];
    size_t wtot = 0, woff[8];
    for (int l = 0; l < 8; ++l) {
        Kreal[l] = CIN[l] * 9;
        Kpad[l]  = (Kreal[l] + 31) & ~31;
        woff[l]  = wtot;
        wtot    += (size_t)COUT[l] * Kpad[l];
    }

    unsigned short* wbf   = (unsigned short*)carve(wtot * 2);
    unsigned short* bufP  = (unsigned short*)carve((size_t)4194304 * 2);
    unsigned short* bufQ  = (unsigned short*)carve((size_t)4194304 * 2);
    float* featX          = (float*)carve((size_t)256 * 4096 * 4);
    float* featY          = (float*)carve((size_t)256 * 4096 * 4);
    float* mu             = (float*)carve(256 * 4);
    unsigned short* xnT   = (unsigned short*)carve((size_t)4096 * 256 * 2);
    unsigned short* ynB   = (unsigned short*)carve((size_t)256 * 4096 * 2);
    float* cosm           = (float*)carve((size_t)4096 * 4096 * 4);
    float* stats          = (float*)carve((size_t)4 * 4096 * 4);
    float* kmax           = (float*)carve((size_t)4096 * 4);

    // ---- convert all conv weights to bf16, zero-padded to Kpad ----
    for (int l = 0; l < 8; ++l) {
        const float* ws = (const float*)d_in[2 + 2 * l];
        int n = COUT[l] * Kpad[l];
        convert_weights_kernel<<<(n + 255) / 256, 256, 0, stream>>>(
            ws, wbf + woff[l], COUT[l], Kreal[l], Kpad[l]);
    }

    // ---- VGG feature extraction for both images ----
    for (int im = 0; im < 2; ++im) {
        const float* img = (const float*)d_in[im];
        float* feat = (im == 0) ? featX : featY;

        preprocess_kernel<<<(196608 + 255) / 256, 256, 0, stream>>>(img, bufP, 196608);

        unsigned short* src = bufP;
        unsigned short* dst = bufQ;
        for (int l = 0; l < 8; ++l) {
            const int side = SIDE[l];
            const int M = COUT[l], N = side * side;
            const float* bias = (const float*)d_in[3 + 2 * l];
            float* outF = (l == 7) ? feat : nullptr;
            dim3 grid(N / 128, M / 64);
            wmma_gemm_kernel<1><<<grid, 256, 0, stream>>>(
                wbf + woff[l], src, bias, outF, dst,
                M, N, Kreal[l], Kpad[l], side, side, 1);
            unsigned short* tp = src; src = dst; dst = tp;

            if (l == 1 || l == 3) {
                int C = COUT[l], OH = side / 2, OW = side / 2;
                int tot = C * OH * OW;
                maxpool_kernel<<<(tot + 255) / 256, 256, 0, stream>>>(src, dst, C, OH, OW);
                tp = src; src = dst; dst = tp;
            }
        }
    }

    // ---- feature centering + normalization ----
    channel_mean_kernel<<<256, 256, 0, stream>>>(featY, mu);
    normalize_kernel<<<4096, 256, 0, stream>>>(featX, mu, xnT, 1);
    normalize_kernel<<<4096, 256, 0, stream>>>(featY, mu, ynB, 0);

    // ---- cosine GEMM: cos[4096][4096] = xnT(4096x256) * ynB(256x4096) ----
    {
        dim3 grid(4096 / 128, 4096 / 64);
        wmma_gemm_kernel<0><<<grid, 256, 0, stream>>>(
            xnT, ynB, nullptr, cosm, nullptr, 4096, 4096, 256, 256, 0, 0, 0);
    }

    // ---- CX reductions + final loss ----
    cx_rowstats_kernel<<<4096, 256, 0, stream>>>(cosm, stats);
    cx_rowmax_kernel<<<4096, 256, 0, stream>>>(cosm, stats, kmax);
    cx_final_kernel<<<1, 256, 0, stream>>>(kmax, (float*)d_out);
}
